// FusedLayer_73770358276767
// MI455X (gfx1250) — compile-verified, matrix operand order checked by reading, not run on hardware
//
#include <hip/hip_runtime.h>
#include <hip/hip_bf16.h>

// ---------------------------------------------------------------------------
// Problem constants (match reference)
// ---------------------------------------------------------------------------
#define Cdim   128
#define Hh     8
#define Fdim   128
#define FFdim  512
#define Nn     50000
#define Ee     400000
#define ETt    4096
#define Ss     16
#define FUSEDd 384
#define AVG_DEG_LOG 2.1972245773362196f   // log(9)

// Param offsets (f32 elements) inside d_in[3], insertion order of setup_inputs
#define OFF_Wqkv  0
#define OFF_bqkv  49152
#define OFF_Wo    49536
#define OFF_bo    65920
#define OFF_g1    66048
#define OFF_b1    66176
#define OFF_Wf1   66304
#define OFF_bf1   131840
#define OFF_Wf2   132352
#define OFF_bf2   197888
#define OFF_g2    198016
#define OFF_b2    198144
#define OFF_gtab  198272
#define OFF_btab  198400
#define OFF_We    198528
#define OFF_be    214912
#define OFF_Wpre  215040
#define OFF_bpre  264192
#define OFF_Wpost 264320
#define OFF_bpost 477312
#define OFF_Wlin  477440
#define OFF_blin  493824
#define OFF_gbn   493952
#define OFF_bbn   494080
// Weu1/beu1/Weu2/beu2 dead (494208..560000)
#define OFF_gf0   560000
#define OFF_bf0   560384
#define OFF_Wu1   560768
#define OFF_bu1   1150592
#define OFF_Wu2   1152128
#define OFF_bu2   3511424
#define OFF_Wu3   3512960
#define OFF_bu3   4102784
#define OFF_gfn   4103168
#define OFF_bfn   4103552
#define P_TOTAL   4103936

// ---------------------------------------------------------------------------
// WMMA types (CDNA5 gfx1250, wave32)
// ---------------------------------------------------------------------------
typedef __bf16 v16bf __attribute__((ext_vector_type(16)));
typedef float  v8f   __attribute__((ext_vector_type(8)));

union FragB { v16bf v; unsigned u[8]; };
union FragC { v8f   v; float    f[8]; };

#define LDT 40   // LDS row stride in bf16: 80B row => 20-bank stride, conflict-free, 16B aligned

// Per-lane K offsets for 16-bit A/B fragments (ISA 7.12.2: 16-bit A 16x32 table)
__device__ __forceinline__ void frag_koff(int half, int* koff) {
#pragma unroll
  for (int j = 0; j < 8; j++)
    koff[j] = (j < 4) ? (2 * j + 8 * half) : (16 + 2 * (j - 4) + 8 * half);
}

// ---------------------------------------------------------------------------
// Generic tiled GEMM:  C[M,N] = act(A[M,K] @ W[N,K]^T + bias)
// A: f32 (converted to bf16 in LDS) or bf16. W: bf16, row stride ldw.
// Block tile 64x64, 8 waves, each wave one 16-row tile x two 16-col tiles.
// Requires: K % 32 == 0, N % 64 == 0. M tail guarded.
// ---------------------------------------------------------------------------
template <typename TA, int ACT>  // ACT: 0 none, 1 relu, 2 leaky(0.01)
__global__ __launch_bounds__(256) void gemm_kernel(
    const TA* __restrict__ A, const __bf16* __restrict__ W, int ldw,
    const float* __restrict__ bias, float* __restrict__ C,
    int M, int N, int K) {
  __shared__ __bf16 sA[64 * LDT];
  __shared__ __bf16 sW[64 * LDT];
  const int t    = threadIdx.x;
  const int lane = t & 31;
  const int w    = t >> 5;
  const int bm   = blockIdx.y * 64;
  const int bn   = blockIdx.x * 64;
  const int rm   = (w >> 1) * 16;   // 0,16,32,48
  const int cn   = (w & 1) * 32;    // 0 or 32
  const int half = lane >> 4;
  const int l15  = lane & 15;

  FragC c0, c1;
#pragma unroll
  for (int i = 0; i < 8; i++) { c0.f[i] = 0.f; c1.f[i] = 0.f; }
  int koff[8];
  frag_koff(half, koff);

  for (int k0 = 0; k0 < K; k0 += 32) {
    __syncthreads();
    if constexpr (sizeof(TA) == 4) {  // f32 input -> convert to bf16
#pragma unroll
      for (int i = 0; i < 2; i++) {
        int q = t * 2 + i;                 // 0..511 float4 chunks
        int r = q >> 3, c4 = (q & 7) * 4;
        int gr = bm + r;
        float4 val = make_float4(0.f, 0.f, 0.f, 0.f);
        if (gr < M) val = *(const float4*)((const float*)A + (size_t)gr * K + k0 + c4);
        __bf16* dst = &sA[r * LDT + c4];
        dst[0] = (__bf16)val.x; dst[1] = (__bf16)val.y;
        dst[2] = (__bf16)val.z; dst[3] = (__bf16)val.w;
      }
    } else {  // bf16 input
      int q = t;                           // 0..255 uint4 chunks
      int r = q >> 2, c8 = (q & 3) * 8;
      int gr = bm + r;
      uint4 val = make_uint4(0u, 0u, 0u, 0u);
      if (gr < M) val = *(const uint4*)((const __bf16*)A + (size_t)gr * K + k0 + c8);
      *(uint4*)&sA[r * LDT + c8] = val;
    }
    {  // weight tile (bn+r < N guaranteed, N % 64 == 0)
      int q = t;
      int r = q >> 2, c8 = (q & 3) * 8;
      *(uint4*)&sW[r * LDT + c8] = *(const uint4*)(W + (size_t)(bn + r) * ldw + k0 + c8);
    }
    __syncthreads();

    FragB a, b0, b1;
#pragma unroll
    for (int j = 0; j < 8; j++) {
      a.u[j]  = *(const unsigned*)&sA[(rm + l15) * LDT + koff[j]];
      b0.u[j] = *(const unsigned*)&sW[(cn + l15) * LDT + koff[j]];
      b1.u[j] = *(const unsigned*)&sW[(cn + 16 + l15) * LDT + koff[j]];
    }
    c0.v = __builtin_amdgcn_wmma_f32_16x16x32_bf16(false, a.v, false, b0.v, (short)0, c0.v, false, false);
    c1.v = __builtin_amdgcn_wmma_f32_16x16x32_bf16(false, a.v, false, b1.v, (short)0, c1.v, false, false);
  }

#pragma unroll
  for (int tt = 0; tt < 2; tt++) {
    FragC& c = tt ? c1 : c0;
    int col = bn + cn + tt * 16 + l15;
    float bv = bias ? bias[col] : 0.f;
#pragma unroll
    for (int r = 0; r < 8; r++) {
      int row = bm + rm + r + 8 * half;
      if (row < M) {
        float v = c.f[r] + bv;
        if constexpr (ACT == 1) v = fmaxf(v, 0.f);
        if constexpr (ACT == 2) v = (v > 0.f) ? v : 0.01f * v;
        C[(size_t)row * N + col] = v;
      }
    }
  }
}

// ---------------------------------------------------------------------------
// Float atomic max/min via sign-aware int atomics (works with +-inf init)
// ---------------------------------------------------------------------------
__device__ __forceinline__ void atomicMaxF(float* a, float v) {
  if (v >= 0.f) atomicMax((int*)a, __float_as_int(v));
  else          atomicMin((unsigned*)a, __float_as_uint(v));
}
__device__ __forceinline__ void atomicMinF(float* a, float v) {
  if (v >= 0.f) atomicMin((int*)a, __float_as_int(v));
  else          atomicMax((unsigned*)a, __float_as_uint(v));
}

// ---------------------------------------------------------------------------
// Fused PNA edge kernel:
//   m[e,:] = edge_attr[e,:] @ M3^T + b3 + P_dst[dst[e],:] + P_src[src[e],:]
// then scatter-aggregate (sum, sumsq, max, min, count) by dst via atomics.
// Block = 64 edges x 128 channels, WMMA for the GEMM part.
// ---------------------------------------------------------------------------
__global__ __launch_bounds__(256) void edge_pna_kernel(
    const float* __restrict__ edge_attr, const __bf16* __restrict__ M3,
    const float* __restrict__ b3, const int* __restrict__ eidx,
    const float* __restrict__ Pdst, const float* __restrict__ Psrc,
    float* __restrict__ sumb, float* __restrict__ sqb,
    float* __restrict__ mxb, float* __restrict__ mnb, float* __restrict__ cntb,
    int E) {
  __shared__ __bf16 sA[64 * LDT];
  __shared__ __bf16 sW[128 * LDT];
  __shared__ int sDst[64], sSrc[64];
  const int t = threadIdx.x, lane = t & 31, w = t >> 5;
  const int be = blockIdx.x * 64;
  const int rm = (w & 3) * 16;        // edge row tile
  const int cbase = (w >> 2) * 64;    // channel base (2 waves x 64 ch)
  const int half = lane >> 4, l15 = lane & 15;
  if (t < 64) {
    int ge = be + t;
    sSrc[t] = (ge < E) ? eidx[ge] : 0;        // edge_index[0] = src
    sDst[t] = (ge < E) ? eidx[E + ge] : 0;    // edge_index[1] = dst
  }
  FragC c[4];
#pragma unroll
  for (int i = 0; i < 4; i++)
#pragma unroll
    for (int j = 0; j < 8; j++) c[i].f[j] = 0.f;
  int koff[8];
  frag_koff(half, koff);

  for (int k0 = 0; k0 < 128; k0 += 32) {
    __syncthreads();
#pragma unroll
    for (int i = 0; i < 2; i++) {           // edge_attr tile 64x32 f32 -> bf16
      int q = t * 2 + i;
      int r = q >> 3, c4 = (q & 7) * 4;
      int gr = be + r;
      float4 val = make_float4(0.f, 0.f, 0.f, 0.f);
      if (gr < E) val = *(const float4*)(edge_attr + (size_t)gr * 128 + k0 + c4);
      __bf16* dst = &sA[r * LDT + c4];
      dst[0] = (__bf16)val.x; dst[1] = (__bf16)val.y;
      dst[2] = (__bf16)val.z; dst[3] = (__bf16)val.w;
    }
#pragma unroll
    for (int i = 0; i < 2; i++) {           // M3 tile 128x32 bf16
      int q = t + 256 * i;
      int r = q >> 2, c8 = (q & 3) * 8;
      *(uint4*)&sW[r * LDT + c8] = *(const uint4*)(M3 + (size_t)r * 128 + k0 + c8);
    }
    __syncthreads();
    FragB a;
#pragma unroll
    for (int j = 0; j < 8; j++)
      a.u[j] = *(const unsigned*)&sA[(rm + l15) * LDT + koff[j]];
#pragma unroll
    for (int tt = 0; tt < 4; tt++) {
      FragB b;
#pragma unroll
      for (int j = 0; j < 8; j++)
        b.u[j] = *(const unsigned*)&sW[(cbase + 16 * tt + l15) * LDT + koff[j]];
      c[tt].v = __builtin_amdgcn_wmma_f32_16x16x32_bf16(false, a.v, false, b.v, (short)0, c[tt].v, false, false);
    }
  }

  // epilogue: add b3 + gathered node projections, scatter atomics by dst
#pragma unroll
  for (int tt = 0; tt < 4; tt++) {
    int ch = cbase + 16 * tt + l15;
    float b3v = b3[ch];
#pragma unroll
    for (int r = 0; r < 8; r++) {
      int er = rm + r + 8 * half;           // local edge row
      if (be + er < E) {
        int nd = sDst[er], ns = sSrc[er];
        float m = c[tt].f[r] + b3v + Pdst[(size_t)nd * 128 + ch] + Psrc[(size_t)ns * 128 + ch];
        size_t off = (size_t)nd * 128 + ch;
        atomicAdd(&sumb[off], m);
        atomicAdd(&sqb[off], m * m);
        atomicMaxF(&mxb[off], m);
        atomicMinF(&mnb[off], m);
      }
    }
  }
  if (cbase == 0 && l15 == 0) {             // in-degree count: once per edge
#pragma unroll
    for (int r = 0; r < 8; r++) {
      int er = rm + r + 8 * half;
      if (be + er < E) atomicAdd(&cntb[sDst[er]], 1.0f);
    }
  }
}

// ---------------------------------------------------------------------------
// Small VALU kernels
// ---------------------------------------------------------------------------
__global__ void cvt_params_kernel(const float* __restrict__ in, __bf16* __restrict__ out, int n) {
  int i = blockIdx.x * 256 + threadIdx.x;
  if (i < n) out[i] = (__bf16)in[i];
}

// M3 = Wpre[:,256:384] @ We  (so  Pe = edge_attr @ M3^T);  b3 = Wpre[:,256:]@be + bpre
__global__ void fold_kernel(const float* __restrict__ p, __bf16* __restrict__ M3, float* __restrict__ b3) {
  int idx = blockIdx.x * 256 + threadIdx.x;
  if (idx < 128 * 128) {
    int f = idx >> 7, j = idx & 127;
    float acc = 0.f;
    for (int k = 0; k < 128; k++)
      acc += p[OFF_Wpre + f * 384 + 256 + k] * p[OFF_We + k * 128 + j];
    M3[idx] = (__bf16)acc;
  }
  if (idx < 128) {
    float acc = p[OFF_bpre + idx];
    for (int k = 0; k < 128; k++)
      acc += p[OFF_Wpre + idx * 384 + 256 + k] * p[OFF_be + k];
    b3[idx] = acc;
  }
}

__global__ void init_agg_kernel(float* sumb, float* sqb, float* mxb, float* mnb,
                                float* cntb, float* bnsum, float* bnsq, int n) {
  int i = blockIdx.x * 256 + threadIdx.x;
  if (i < n) {
    sumb[i] = 0.f; sqb[i] = 0.f;
    mxb[i] = __int_as_float(0xff800000);   // -inf
    mnb[i] = __int_as_float(0x7f800000);   // +inf
  }
  if (i < Nn) cntb[i] = 0.f;
  if (i < 128) { bnsum[i] = 0.f; bnsq[i] = 0.f; }
}

// Attention for query position s==0 only (per (batch, head), head_dim=16)
__global__ __launch_bounds__(128) void attn_kernel(const float* __restrict__ qkv,
                                                   float* __restrict__ o, int B) {
  int idx = blockIdx.x * 128 + threadIdx.x;
  if (idx >= B * Hh) return;
  int b = idx >> 3, h = idx & 7;
  const float* base = qkv + (size_t)b * Ss * 384;
  float q0[16];
#pragma unroll
  for (int d = 0; d < 16; d++) q0[d] = base[h * 16 + d];
  float lg[16], mx = -1e30f;
  for (int s = 0; s < 16; s++) {
    const float* kr = base + s * 384 + 128 + h * 16;
    float acc = 0.f;
#pragma unroll
    for (int d = 0; d < 16; d++) acc += q0[d] * kr[d];
    acc *= 0.25f;                 // 1/sqrt(16)
    lg[s] = acc; mx = fmaxf(mx, acc);
  }
  float den = 0.f;
  for (int s = 0; s < 16; s++) { lg[s] = __expf(lg[s] - mx); den += lg[s]; }
  float inv = 1.f / den;
  float ov[16];
#pragma unroll
  for (int d = 0; d < 16; d++) ov[d] = 0.f;
  for (int s = 0; s < 16; s++) {
    float pp = lg[s] * inv;
    const float* vr = base + s * 384 + 256 + h * 16;
#pragma unroll
    for (int d = 0; d < 16; d++) ov[d] += pp * vr[d];
  }
#pragma unroll
  for (int d = 0; d < 16; d++) o[(size_t)b * 128 + h * 16 + d] = ov[d];
}

// LayerNorm: out = LN(in1 + in2?), one wave per row, F = PER*32
template <int PER>
__global__ __launch_bounds__(256) void ln_kernel(
    const float* __restrict__ in1, int s1, const float* __restrict__ in2, int s2,
    const float* __restrict__ g, const float* __restrict__ b,
    float* __restrict__ out, int rows) {
  int wv = threadIdx.x >> 5, lane = threadIdx.x & 31;
  int row = blockIdx.x * 8 + wv;
  if (row >= rows) return;
  const int F = PER * 32;
  float v[PER], s = 0.f;
#pragma unroll
  for (int i = 0; i < PER; i++) {
    float x = in1[(size_t)row * s1 + lane + 32 * i];
    if (in2) x += in2[(size_t)row * s2 + lane + 32 * i];
    v[i] = x; s += x;
  }
#pragma unroll
  for (int o = 16; o > 0; o >>= 1) s += __shfl_xor(s, o, 32);
  float mu = s / F, q = 0.f;
#pragma unroll
  for (int i = 0; i < PER; i++) { float d = v[i] - mu; q += d * d; }
#pragma unroll
  for (int o = 16; o > 0; o >>= 1) q += __shfl_xor(q, o, 32);
  float rs = rsqrtf(q / F + 1e-5f);
#pragma unroll
  for (int i = 0; i < PER; i++) {
    int cc = lane + 32 * i;
    out[(size_t)row * F + cc] = (v[i] - mu) * rs * g[cc] + b[cc];
  }
}

// out = (X + LN(y)) / 2    (final fuse output)
template <int PER>
__global__ __launch_bounds__(256) void ln_final_kernel(
    const float* __restrict__ y, const float* __restrict__ X,
    const float* __restrict__ g, const float* __restrict__ b,
    float* __restrict__ out, int rows) {
  int wv = threadIdx.x >> 5, lane = threadIdx.x & 31;
  int row = blockIdx.x * 8 + wv;
  if (row >= rows) return;
  const int F = PER * 32;
  float v[PER], s = 0.f;
#pragma unroll
  for (int i = 0; i < PER; i++) { v[i] = y[(size_t)row * F + lane + 32 * i]; s += v[i]; }
#pragma unroll
  for (int o = 16; o > 0; o >>= 1) s += __shfl_xor(s, o, 32);
  float mu = s / F, q = 0.f;
#pragma unroll
  for (int i = 0; i < PER; i++) { float d = v[i] - mu; q += d * d; }
#pragma unroll
  for (int o = 16; o > 0; o >>= 1) q += __shfl_xor(q, o, 32);
  float rs = rsqrtf(q / F + 1e-5f);
#pragma unroll
  for (int i = 0; i < PER; i++) {
    int cc = lane + 32 * i;
    float ln = (v[i] - mu) * rs * g[cc] + b[cc];
    out[(size_t)row * F + cc] = (X[(size_t)row * F + cc] + ln) * 0.5f;
  }
}

// Aggregator finalize -> X13 [N, 1664] bf16 = [x, mean,max,min,std, *s1, *s2]
__global__ void finalize_kernel(const float* __restrict__ x_gnn,
                                const float* __restrict__ sumb, const float* __restrict__ sqb,
                                const float* __restrict__ mxb, const float* __restrict__ mnb,
                                const float* __restrict__ cntb, __bf16* __restrict__ X13) {
  int idx = blockIdx.x * 256 + threadIdx.x;
  if (idx >= Nn * 128) return;
  int n = idx >> 7, ch = idx & 127;
  float cnt = cntb[n];
  float denom = fmaxf(cnt, 1.f);
  float mean = sumb[idx] / denom;
  float m2 = sqb[idx] / denom;
  float sd = sqrtf(fmaxf(m2 - mean * mean, 0.f) + 1e-5f);
  float mx = (cnt > 0.f) ? mxb[idx] : 0.f;
  float mn = (cnt > 0.f) ? mnb[idx] : 0.f;
  float logd = logf(denom + 1.f);
  float s1 = logd / AVG_DEG_LOG, s2 = AVG_DEG_LOG / logd;
  __bf16* row = X13 + (size_t)n * 1664;
  row[ch]        = (__bf16)x_gnn[idx];
  row[128 + ch]  = (__bf16)mean;       row[256 + ch]  = (__bf16)mx;
  row[384 + ch]  = (__bf16)mn;         row[512 + ch]  = (__bf16)sd;
  row[640 + ch]  = (__bf16)(mean*s1);  row[768 + ch]  = (__bf16)(mx*s1);
  row[896 + ch]  = (__bf16)(mn*s1);    row[1024 + ch] = (__bf16)(sd*s1);
  row[1152 + ch] = (__bf16)(mean*s2);  row[1280 + ch] = (__bf16)(mx*s2);
  row[1408 + ch] = (__bf16)(mn*s2);    row[1536 + ch] = (__bf16)(sd*s2);
}

// BatchNorm stats: per-channel sum/sumsq over rows, low-contention atomics
__global__ __launch_bounds__(256) void bn_stats_kernel(const float* __restrict__ t2,
                                                       float* __restrict__ bnsum,
                                                       float* __restrict__ bnsq, int N) {
  int c = threadIdx.x & 127, sub = threadIdx.x >> 7;
  float s = 0.f, q = 0.f;
  for (int r = blockIdx.x * 2 + sub; r < N; r += gridDim.x * 2) {
    float v = t2[(size_t)r * 128 + c];
    s += v; q += v * v;
  }
  __shared__ float ls[256], lq[256];
  ls[threadIdx.x] = s; lq[threadIdx.x] = q;
  __syncthreads();
  if (sub == 0) {
    atomicAdd(&bnsum[c], s + ls[128 + c]);
    atomicAdd(&bnsq[c],  q + lq[128 + c]);
  }
}

// x_g = (x_gnn + relu(BN(t2))) / 2
__global__ void bn_apply_kernel(const float* __restrict__ t2, const float* __restrict__ x_gnn,
                                const float* __restrict__ bnsum, const float* __restrict__ bnsq,
                                const float* __restrict__ gbn, const float* __restrict__ bbn,
                                float* __restrict__ x_g) {
  int idx = blockIdx.x * 256 + threadIdx.x;
  if (idx >= Nn * 128) return;
  int ch = idx & 127;
  float invN = 1.f / (float)Nn;
  float mu = bnsum[ch] * invN;
  float var = fmaxf(bnsq[ch] * invN - mu * mu, 0.f);
  float o = (t2[idx] - mu) * rsqrtf(var + 1e-5f) * gbn[ch] + bbn[ch];
  x_g[idx] = (x_gnn[idx] + fmaxf(o, 0.f)) * 0.5f;
}

// X = concat(x_cls, x_g[tgt0], x_g[tgt1])  [ET, 384]
__global__ void fuse_build_kernel(const float* __restrict__ x_cls, const float* __restrict__ x_g,
                                  const int* __restrict__ tgt, float* __restrict__ X) {
  int idx = blockIdx.x * 256 + threadIdx.x;
  if (idx >= ETt * FUSEDd) return;
  int t = idx / FUSEDd, c = idx - t * FUSEDd;
  float v;
  if (c < 128)       v = x_cls[(size_t)t * 128 + c];
  else if (c < 256)  v = x_g[(size_t)tgt[t] * 128 + (c - 128)];
  else               v = x_g[(size_t)tgt[ETt + t] * 128 + (c - 256)];
  X[idx] = v;
}

// ---------------------------------------------------------------------------
// Host launcher
// ---------------------------------------------------------------------------
extern "C" void kernel_launch(void* const* d_in, const int* in_sizes, int n_in,
                              void* d_out, int out_size, void* d_ws, size_t ws_size,
                              hipStream_t stream) {
  const float* x_tab     = (const float*)d_in[0];   // [4096,16,128]
  const float* x_gnn     = (const float*)d_in[1];   // [50000,128]
  const float* edge_attr = (const float*)d_in[2];   // [400000,128]
  const float* p         = (const float*)d_in[3];   // flat params (insertion order)
  const int*   eidx      = (const int*)d_in[4];     // [2,400000]
  const int*   tgt       = (const int*)d_in[5];     // [2,4096]
  float*       out       = (float*)d_out;           // [4096,384]
  (void)in_sizes; (void)n_in; (void)out_size; (void)ws_size;

  char* base = (char*)d_ws;
  size_t cur = 0;
  auto alloc = [&](size_t bytes) -> void* {
    void* r = base + cur;
    cur = (cur + bytes + 255) & ~(size_t)255;
    return r;
  };

  // ---- persistent region ----
  __bf16* pb16  = (__bf16*)alloc((size_t)P_TOTAL * 2);
  __bf16* M3    = (__bf16*)alloc(128 * 128 * 2);
  float*  b3    = (float*) alloc(128 * 4);
  float*  x_cls = (float*) alloc((size_t)ETt * 128 * 4);
  float*  x_g   = (float*) alloc((size_t)Nn * 128 * 4);
  float*  Xf    = (float*) alloc((size_t)ETt * FUSEDd * 4);
  float*  bnsum = (float*) alloc(128 * 4);
  float*  bnsq  = (float*) alloc(128 * 4);
  size_t scratch0 = cur;

  // ---- tab-phase scratch ----
  float* qkv   = (float*)alloc((size_t)ETt * Ss * 384 * 4);
  float* o_    = (float*)alloc((size_t)ETt * 128 * 4);
  float* oproj = (float*)alloc((size_t)ETt * 128 * 4);
  float* hbuf  = (float*)alloc((size_t)ETt * 128 * 4);
  float* ff1   = (float*)alloc((size_t)ETt * 512 * 4);
  float* ff2   = (float*)alloc((size_t)ETt * 128 * 4);
  float* h2    = (float*)alloc((size_t)ETt * 128 * 4);

  // ---- gnn-phase scratch (aliases tab scratch; phases are serialized) ----
  cur = scratch0;
  float*  Pdst = (float*)alloc((size_t)Nn * 128 * 4);
  float*  Psrc = (float*)alloc((size_t)Nn * 128 * 4);
  float*  sumb = (float*)alloc((size_t)Nn * 128 * 4);
  float*  sqb  = (float*)alloc((size_t)Nn * 128 * 4);
  float*  mxb  = (float*)alloc((size_t)Nn * 128 * 4);
  float*  mnb  = (float*)alloc((size_t)Nn * 128 * 4);
  float*  cntb = (float*)alloc((size_t)Nn * 4);
  __bf16* X13  = (__bf16*)alloc((size_t)Nn * 1664 * 2);
  float*  t1   = (float*)alloc((size_t)Nn * 128 * 4);
  float*  t2   = (float*)alloc((size_t)Nn * 128 * 4);

  // ---- fuse-phase scratch (aliases again) ----
  cur = scratch0;
  float* y0 = (float*)alloc((size_t)ETt * FUSEDd * 4);
  float* y1 = (float*)alloc((size_t)ETt * 1536 * 4);
  float* y2 = (float*)alloc((size_t)ETt * 1536 * 4);
  float* y3 = (float*)alloc((size_t)ETt * FUSEDd * 4);

  // ================= prep =================
  cvt_params_kernel<<<(P_TOTAL + 255) / 256, 256, 0, stream>>>(p, pb16, P_TOTAL);
  fold_kernel<<<64, 256, 0, stream>>>(p, M3, b3);
  init_agg_kernel<<<(Nn * 128 + 255) / 256, 256, 0, stream>>>(sumb, sqb, mxb, mnb, cntb, bnsum, bnsq, Nn * 128);

  // ================= tab branch (only token 0 needed post-attention) =======
  // qkv = x_tab @ Wqkv^T + bqkv   [65536, 384]
  gemm_kernel<float, 0><<<dim3(384 / 64, (ETt * Ss) / 64), 256, 0, stream>>>(
      x_tab, pb16 + OFF_Wqkv, 128, p + OFF_bqkv, qkv, ETt * Ss, 384, 128);
  attn_kernel<<<(ETt * Hh + 127) / 128, 128, 0, stream>>>(qkv, o_, ETt);
  gemm_kernel<float, 0><<<dim3(128 / 64, ETt / 64), 256, 0, stream>>>(
      o_, pb16 + OFF_Wo, 128, p + OFF_bo, oproj, ETt, 128, 128);
  ln_kernel<4><<<ETt / 8, 256, 0, stream>>>(x_tab, Ss * 128, oproj, 128,
                                            p + OFF_g1, p + OFF_b1, hbuf, ETt);
  gemm_kernel<float, 1><<<dim3(512 / 64, ETt / 64), 256, 0, stream>>>(
      hbuf, pb16 + OFF_Wf1, 128, p + OFF_bf1, ff1, ETt, 512, 128);
  gemm_kernel<float, 0><<<dim3(128 / 64, ETt / 64), 256, 0, stream>>>(
      ff1, pb16 + OFF_Wf2, 512, p + OFF_bf2, ff2, ETt, 128, 512);
  ln_kernel<4><<<ETt / 8, 256, 0, stream>>>(hbuf, 128, ff2, 128,
                                            p + OFF_g2, p + OFF_b2, h2, ETt);
  ln_kernel<4><<<ETt / 8, 256, 0, stream>>>(h2, 128, nullptr, 0,
                                            p + OFF_gtab, p + OFF_btab, x_cls, ETt);

  // ================= PNA branch =================
  // node projections (Wpre split columns; ldw = 384)
  gemm_kernel<float, 0><<<dim3(2, (Nn + 63) / 64), 256, 0, stream>>>(
      x_gnn, pb16 + OFF_Wpre, 384, nullptr, Pdst, Nn, 128, 128);
  gemm_kernel<float, 0><<<dim3(2, (Nn + 63) / 64), 256, 0, stream>>>(
      x_gnn, pb16 + OFF_Wpre + 128, 384, nullptr, Psrc, Nn, 128, 128);
  edge_pna_kernel<<<(Ee + 63) / 64, 256, 0, stream>>>(
      edge_attr, M3, b3, eidx, Pdst, Psrc, sumb, sqb, mxb, mnb, cntb, Ee);
  finalize_kernel<<<(Nn * 128 + 255) / 256, 256, 0, stream>>>(
      x_gnn, sumb, sqb, mxb, mnb, cntb, X13);
  gemm_kernel<__bf16, 0><<<dim3(2, (Nn + 63) / 64), 256, 0, stream>>>(
      X13, pb16 + OFF_Wpost, 1664, p + OFF_bpost, t1, Nn, 128, 1664);
  gemm_kernel<float, 0><<<dim3(2, (Nn + 63) / 64), 256, 0, stream>>>(
      t1, pb16 + OFF_Wlin, 128, p + OFF_blin, t2, Nn, 128, 128);
  bn_stats_kernel<<<256, 256, 0, stream>>>(t2, bnsum, bnsq, Nn);
  bn_apply_kernel<<<(Nn * 128 + 255) / 256, 256, 0, stream>>>(
      t2, x_gnn, bnsum, bnsq, p + OFF_gbn, p + OFF_bbn, x_g);

  // ================= fuse =================
  fuse_build_kernel<<<(ETt * FUSEDd + 255) / 256, 256, 0, stream>>>(x_cls, x_g, tgt, Xf);
  ln_kernel<12><<<ETt / 8, 256, 0, stream>>>(Xf, FUSEDd, nullptr, 0,
                                             p + OFF_gf0, p + OFF_bf0, y0, ETt);
  gemm_kernel<float, 2><<<dim3(1536 / 64, ETt / 64), 256, 0, stream>>>(
      y0, pb16 + OFF_Wu1, 384, p + OFF_bu1, y1, ETt, 1536, 384);
  gemm_kernel<float, 2><<<dim3(1536 / 64, ETt / 64), 256, 0, stream>>>(
      y1, pb16 + OFF_Wu2, 1536, p + OFF_bu2, y2, ETt, 1536, 1536);
  gemm_kernel<float, 0><<<dim3(384 / 64, ETt / 64), 256, 0, stream>>>(
      y2, pb16 + OFF_Wu3, 1536, p + OFF_bu3, y3, ETt, 384, 1536);
  ln_final_kernel<12><<<ETt / 8, 256, 0, stream>>>(y3, Xf, p + OFF_gfn, p + OFF_bfn, out, ETt);
}
